// VanillaRNN_64201171140979
// MI455X (gfx1250) — matrix operand order repaired and checked by reading, not compiled, not run on hardware
//
#include <hip/hip_runtime.h>
#include <hip/hip_bf16.h>
#include <math.h>

typedef __bf16 bf16_t;
typedef __attribute__((ext_vector_type(16))) __bf16 v16bf;
typedef __attribute__((ext_vector_type(8)))  __bf16 v8bf;
typedef __attribute__((ext_vector_type(8)))  float  v8f;

#define RNN_B 64
#define RNN_T 2048
#define RNN_V 256
#define RNN_H 1024
#define GRID_BLKS 20   // 16 recurrence blocks + 4 logits/softmax blocks
#define A_BLKS 16

union FragBF {
  v16bf v;
  struct { v8bf lo, hi; } h;
};

// ---------------------------------------------------------------------------
// Grid-wide sense-reversing barrier (agent scope). All GRID_BLKS blocks are
// co-resident (tiny grid), so spinning is safe. Works for any initial `gen`,
// requires count==0 at kernel start (memsetAsync'd every launch).
// ---------------------------------------------------------------------------
__device__ __forceinline__ void grid_barrier(unsigned* cnt, unsigned* gen) {
  __syncthreads();
  if (threadIdx.x == 0) {
    __threadfence();  // make this block's global writes visible device-wide
    unsigned g = __hip_atomic_load(gen, __ATOMIC_RELAXED, __HIP_MEMORY_SCOPE_AGENT);
    unsigned arrived =
        __hip_atomic_fetch_add(cnt, 1u, __ATOMIC_ACQ_REL, __HIP_MEMORY_SCOPE_AGENT);
    if (arrived == GRID_BLKS - 1) {
      __hip_atomic_store(cnt, 0u, __ATOMIC_RELAXED, __HIP_MEMORY_SCOPE_AGENT);
      __hip_atomic_store(gen, g + 1u, __ATOMIC_RELEASE, __HIP_MEMORY_SCOPE_AGENT);
    } else {
      while (__hip_atomic_load(gen, __ATOMIC_ACQUIRE, __HIP_MEMORY_SCOPE_AGENT) == g) {
        __builtin_amdgcn_s_sleep(1);
      }
    }
  }
  __syncthreads();
}

// ---------------------------------------------------------------------------
// Prologue: transpose + convert weights to bf16 so WMMA B-fragments (lane =
// output column, 16 consecutive K values per lane) load contiguously.
//   whhT[n][k] = Whh[k][n]   (1024x1024)
//   whyT[v][k] = Why[k][v]   ( 256x1024)
// ---------------------------------------------------------------------------
__global__ void rnn_prep_weights(const float* __restrict__ Whh,
                                 const float* __restrict__ Why,
                                 bf16_t* __restrict__ whhT,
                                 bf16_t* __restrict__ whyT) {
  int idx = blockIdx.x * 256 + threadIdx.x;
  const int NWHH = RNN_H * RNN_H;
  if (idx < NWHH) {
    int n = idx >> 10, k = idx & (RNN_H - 1);
    whhT[idx] = (bf16_t)Whh[k * RNN_H + n];
  } else if (idx < NWHH + RNN_V * RNN_H) {
    int j = idx - NWHH;
    int v = j >> 10, k = j & (RNN_H - 1);
    whyT[j] = (bf16_t)Why[k * RNN_V + v];
  }
}

// ---------------------------------------------------------------------------
// Persistent cooperative RNN kernel.
//  Blocks 0..15 ("A"): h_{t+1} = tanh(Wxh[tok_t] + h_t @ Whh + bh)
//      256 output tiles (4 Mtiles x 64 Ntiles), 16 per block, 2 per wave,
//      each tile = 32 x v_wmma_f32_16x16x32_bf16 over K=1024.
//  Blocks 16..19 ("B"): at iteration t compute CE loss of step t-1 from the
//      just-finished h (other buffer): logits 16x256 via WMMA, then wave32
//      shfl_xor log-softmax + target gather, accumulate into LDS.
//  One grid barrier per timestep pipelines A(t+1) with B(t).
// ---------------------------------------------------------------------------
__global__ void __launch_bounds__(256, 1)
rnn_persistent(const int* __restrict__ tokens, const float* __restrict__ Wxh,
               const float* __restrict__ bh, const float* __restrict__ by,
               const bf16_t* __restrict__ whhT, const bf16_t* __restrict__ whyT,
               bf16_t* __restrict__ hbuf0, bf16_t* __restrict__ hbuf1,
               unsigned* __restrict__ sync, float* __restrict__ out) {
  const int tid    = threadIdx.x;
  const int lane   = tid & 31;
  const int wave   = tid >> 5;
  const int laneLo = lane & 15;   // WMMA: A row / B column / D column
  const int laneHi = lane >> 4;   // WMMA: K-half select / D row-half select
  const int blk    = blockIdx.x;

  __shared__ float lds_logits[16][RNN_V];  // 16KB (B-blocks)
  __shared__ float lds_loss[16];

  unsigned* cnt = sync;
  unsigned* gen = sync + 1;

  if (tid < 16) lds_loss[tid] = 0.0f;
  __syncthreads();

  const bool isA = (blk < A_BLKS);

  // Phase-A static tile assignment: two N-adjacent tiles per wave, same Mtile.
  const int tileId = blk * 16 + wave * 2;
  const int mt  = tileId >> 6;      // 0..3
  const int nt0 = tileId & 63;      // even
  // Phase-B: block owns 16 sequences.
  const int gB = blk - A_BLKS;      // 0..3

  for (int t = 0; t < RNN_T - 1; ++t) {
    // Ternary select keeps both candidates provably global (addrspace(1)):
    // the backend emits global_load_b128, not flat_load_b128 (no DScnt tie).
    const bf16_t* hcur = (t & 1) ? hbuf1 : hbuf0;  // hbuf0 zeroed: h0 == 0
    bf16_t*       hnxt = (t & 1) ? hbuf0 : hbuf1;

    if (isA) {
      v8f acc0 = {}; v8f acc1 = {};
      const int arow = mt * 16 + laneLo;
      const bf16_t* aptr  = hcur + arow * RNN_H + laneHi * 16;
      const bf16_t* bptr0 = whhT + (nt0 * 16 + laneLo) * RNN_H + laneHi * 16;
      const bf16_t* bptr1 = bptr0 + 16 * RNN_H;
      for (int k = 0; k < RNN_H; k += 32) {
        FragBF a, b0, b1;
        a.h.lo  = *(const v8bf*)(aptr + k);
        a.h.hi  = *(const v8bf*)(aptr + k + 8);
        b0.h.lo = *(const v8bf*)(bptr0 + k);
        b0.h.hi = *(const v8bf*)(bptr0 + k + 8);
        b1.h.lo = *(const v8bf*)(bptr1 + k);
        b1.h.hi = *(const v8bf*)(bptr1 + k + 8);
        acc0 = __builtin_amdgcn_wmma_f32_16x16x32_bf16(false, a.v, false, b0.v,
                                                       (short)0, acc0, false, false);
        acc1 = __builtin_amdgcn_wmma_f32_16x16x32_bf16(false, a.v, false, b1.v,
                                                       (short)0, acc1, false, false);
      }
      // Epilogue: + one-hot gather (Wxh[tok]) + bh, tanh, store bf16 h_{t+1}.
      const int col0 = nt0 * 16 + laneLo;
      const int col1 = col0 + 16;
      const float bh0 = bh[col0], bh1 = bh[col1];
#pragma unroll
      for (int i = 0; i < 8; ++i) {
        const int row = mt * 16 + laneHi * 8 + i;       // D: M = i + 8*laneHi
        const int tok = tokens[row * RNN_T + t];
        const float v0 = tanhf(acc0[i] + Wxh[tok * RNN_H + col0] + bh0);
        const float v1 = tanhf(acc1[i] + Wxh[tok * RNN_H + col1] + bh1);
        hnxt[row * RNN_H + col0] = (bf16_t)v0;
        hnxt[row * RNN_H + col1] = (bf16_t)v1;
      }
    } else if (t > 0) {
      // Loss of step s = t-1: uses h_{s+1} == hcur, targets tokens[:, t].
      const int ntB0 = wave * 2, ntB1 = ntB0 + 1;
      v8f c0 = {}; v8f c1 = {};
      const bf16_t* hrow = hcur + (gB * 16 + laneLo) * RNN_H + laneHi * 16;
      const bf16_t* wp0  = whyT + (ntB0 * 16 + laneLo) * RNN_H + laneHi * 16;
      const bf16_t* wp1  = wp0 + 16 * RNN_H;
      for (int k = 0; k < RNN_H; k += 32) {
        FragBF a, b0, b1;
        a.h.lo  = *(const v8bf*)(hrow + k);
        a.h.hi  = *(const v8bf*)(hrow + k + 8);
        b0.h.lo = *(const v8bf*)(wp0 + k);
        b0.h.hi = *(const v8bf*)(wp0 + k + 8);
        b1.h.lo = *(const v8bf*)(wp1 + k);
        b1.h.hi = *(const v8bf*)(wp1 + k + 8);
        c0 = __builtin_amdgcn_wmma_f32_16x16x32_bf16(false, a.v, false, b0.v,
                                                     (short)0, c0, false, false);
        c1 = __builtin_amdgcn_wmma_f32_16x16x32_bf16(false, a.v, false, b1.v,
                                                     (short)0, c1, false, false);
      }
      const float by0 = by[ntB0 * 16 + laneLo];
      const float by1 = by[ntB1 * 16 + laneLo];
#pragma unroll
      for (int i = 0; i < 8; ++i) {
        const int r = laneHi * 8 + i;
        lds_logits[r][ntB0 * 16 + laneLo] = c0[i] + by0;
        lds_logits[r][ntB1 * 16 + laneLo] = c1[i] + by1;
      }
      __syncthreads();
      // Wave32 log-softmax + CE: wave w reduces rows 2w, 2w+1.
#pragma unroll
      for (int rr = 0; rr < 2; ++rr) {
        const int r = wave * 2 + rr;
        float m = -INFINITY;
#pragma unroll
        for (int j = 0; j < 8; ++j) m = fmaxf(m, lds_logits[r][lane + j * 32]);
        for (int off = 16; off; off >>= 1) m = fmaxf(m, __shfl_xor(m, off, 32));
        float s = 0.0f;
#pragma unroll
        for (int j = 0; j < 8; ++j) s += expf(lds_logits[r][lane + j * 32] - m);
        for (int off = 16; off; off >>= 1) s += __shfl_xor(s, off, 32);
        if (lane == 0) {
          const int bidx = gB * 16 + r;
          const int tgt  = tokens[bidx * RNN_T + t];
          lds_loss[r] += m + logf(s) - lds_logits[r][tgt];
        }
      }
    }
    grid_barrier(cnt, gen);
  }

  // Epilogue: loss of the final step s = T-2, using h_{T-1} (in hbuf1).
  if (!isA) {
    const bf16_t* hcur = ((RNN_T - 1) & 1) ? hbuf1 : hbuf0;
    const int ntB0 = wave * 2, ntB1 = ntB0 + 1;
    v8f c0 = {}; v8f c1 = {};
    const bf16_t* hrow = hcur + (gB * 16 + laneLo) * RNN_H + laneHi * 16;
    const bf16_t* wp0  = whyT + (ntB0 * 16 + laneLo) * RNN_H + laneHi * 16;
    const bf16_t* wp1  = wp0 + 16 * RNN_H;
    for (int k = 0; k < RNN_H; k += 32) {
      FragBF a, b0, b1;
      a.h.lo  = *(const v8bf*)(hrow + k);
      a.h.hi  = *(const v8bf*)(hrow + k + 8);
      b0.h.lo = *(const v8bf*)(wp0 + k);
      b0.h.hi = *(const v8bf*)(wp0 + k + 8);
      b1.h.lo = *(const v8bf*)(wp1 + k);
      b1.h.hi = *(const v8bf*)(wp1 + k + 8);
      c0 = __builtin_amdgcn_wmma_f32_16x16x32_bf16(false, a.v, false, b0.v,
                                                   (short)0, c0, false, false);
      c1 = __builtin_amdgcn_wmma_f32_16x16x32_bf16(false, a.v, false, b1.v,
                                                   (short)0, c1, false, false);
    }
    const float by0 = by[ntB0 * 16 + laneLo];
    const float by1 = by[ntB1 * 16 + laneLo];
#pragma unroll
    for (int i = 0; i < 8; ++i) {
      const int r = laneHi * 8 + i;
      lds_logits[r][ntB0 * 16 + laneLo] = c0[i] + by0;
      lds_logits[r][ntB1 * 16 + laneLo] = c1[i] + by1;
    }
    __syncthreads();
#pragma unroll
    for (int rr = 0; rr < 2; ++rr) {
      const int r = wave * 2 + rr;
      float m = -INFINITY;
#pragma unroll
      for (int j = 0; j < 8; ++j) m = fmaxf(m, lds_logits[r][lane + j * 32]);
      for (int off = 16; off; off >>= 1) m = fmaxf(m, __shfl_xor(m, off, 32));
      float s = 0.0f;
#pragma unroll
      for (int j = 0; j < 8; ++j) s += expf(lds_logits[r][lane + j * 32] - m);
      for (int off = 16; off; off >>= 1) s += __shfl_xor(s, off, 32);
      if (lane == 0) {
        const int bidx = gB * 16 + r;
        const int tgt  = tokens[bidx * RNN_T + (RNN_T - 1)];
        lds_loss[r] += m + logf(s) - lds_logits[r][tgt];
      }
    }
    __syncthreads();
    if (tid < 16) out[gB * 16 + tid] = lds_loss[tid];
  }
}

// ---------------------------------------------------------------------------
// Launcher. Inputs (setup_inputs order): tokens[i32], Wxh, Whh, Why, bh, by.
// d_out: float[64]. Workspace layout (bytes):
//   [0,256)                    : barrier count/gen (memset 0 each launch)
//   [256, 256+128K)            : hbuf0 bf16 (memset 0 => h0 = 0)
//   [+128K, +256K)             : hbuf1 bf16
//   [+256K, +256K+2M)          : WhhT bf16 [1024][1024]
//   [+256K+2M, +256K+2.5M)     : WhyT bf16 [256][1024]
// ---------------------------------------------------------------------------
extern "C" void kernel_launch(void* const* d_in, const int* in_sizes, int n_in,
                              void* d_out, int out_size, void* d_ws, size_t ws_size,
                              hipStream_t stream) {
  const int*   tokens = (const int*)d_in[0];
  const float* Wxh    = (const float*)d_in[1];
  const float* Whh    = (const float*)d_in[2];
  const float* Why    = (const float*)d_in[3];
  const float* bh     = (const float*)d_in[4];
  const float* by     = (const float*)d_in[5];
  float* out = (float*)d_out;

  char* ws = (char*)d_ws;
  const size_t HBUF = (size_t)RNN_B * RNN_H * sizeof(bf16_t);  // 128KB
  unsigned* sync  = (unsigned*)ws;
  bf16_t*   hbuf0 = (bf16_t*)(ws + 256);
  bf16_t*   hbuf1 = (bf16_t*)(ws + 256 + HBUF);
  bf16_t*   whhT  = (bf16_t*)(ws + 256 + 2 * HBUF);
  bf16_t*   whyT  = (bf16_t*)(ws + 256 + 2 * HBUF + (size_t)RNN_H * RNN_H * sizeof(bf16_t));

  // Zero barrier state + hbuf0 (h0 == 0). Capturable (async, on stream).
  hipMemsetAsync(d_ws, 0, 256 + HBUF, stream);

  const int prepN = RNN_H * RNN_H + RNN_V * RNN_H;
  rnn_prep_weights<<<(prepN + 255) / 256, 256, 0, stream>>>(Whh, Why, whhT, whyT);

  rnn_persistent<<<GRID_BLKS, 256, 0, stream>>>(tokens, Wxh, bh, by, whhT, whyT,
                                                hbuf0, hbuf1, sync, out);
}
